// Model_17274358464971
// MI455X (gfx1250) — matrix operand order repaired and checked by reading, not compiled, not run on hardware
//
#include <hip/hip_runtime.h>
#include <hip/hip_bf16.h>
#include <math.h>

// Problem dimensions (match reference)
#define BB 4096   // batch
#define DD 1024   // feature dim
#define HH 2048   // hidden
#define CC 1000   // classes
#define CP 1024   // classes padded to multiple of 128
#define TT 819200 // total ragged rows
#define FF 32     // ragged feature width

typedef __attribute__((ext_vector_type(16))) __bf16 v16bf;
typedef __attribute__((ext_vector_type(8)))  float  v8f;
typedef __attribute__((ext_vector_type(4)))  int    v4i;

#ifndef __has_builtin
#define __has_builtin(x) 0
#endif
#if __has_builtin(__builtin_amdgcn_global_load_async_to_lds_b128)
#define HAS_ASYNC_LDS 1
#else
#define HAS_ASYNC_LDS 0
#endif

__device__ __forceinline__ unsigned short f2bf(float f) {
    unsigned int u = __float_as_uint(f);
    unsigned int r = (u + 0x7FFFu + ((u >> 16) & 1u)) >> 16; // round-to-nearest-even
    return (unsigned short)r;
}
__device__ __forceinline__ float bf2f(unsigned short u) {
    return __uint_as_float(((unsigned int)u) << 16);
}

// 16-byte copy global -> LDS, via CDNA5 async-to-LDS when available.
// Builtin signature (from compiler diagnostic): (int4 __device__*, int4 __shared__*, imm, imm)
__device__ __forceinline__ void cp16(unsigned short* lds, const unsigned short* g) {
#if HAS_ASYNC_LDS
    __builtin_amdgcn_global_load_async_to_lds_b128(
        (__attribute__((address_space(1))) v4i*)(g),
        (__attribute__((address_space(3))) v4i*)(lds),
        0 /*offset*/, 0 /*cpol*/);
#else
    *(uint4*)lds = *(const uint4*)g;
#endif
}

__device__ __forceinline__ void wait_async_copies() {
#if HAS_ASYNC_LDS
#if __has_builtin(__builtin_amdgcn_s_wait_asynccnt)
    __builtin_amdgcn_s_wait_asynccnt(0);
#else
    asm volatile("s_wait_asynccnt 0" ::: "memory");
#endif
#endif
}

// ---------------------------------------------------------------- utilities
__global__ void zero_f32(float* p, int n) {
    int i = blockIdx.x * blockDim.x + threadIdx.x;
    if (i < n) p[i] = 0.0f;
}

__global__ void cvt_f32_bf16(const float* __restrict__ src, unsigned short* __restrict__ dst, int n) {
    int i = blockIdx.x * blockDim.x + threadIdx.x;
    if (i < n) dst[i] = f2bf(src[i]);
}

// pad W2 [CC x HH] -> [CP x HH] bf16 (pad rows zero)
__global__ void cvt_w2_pad(const float* __restrict__ w2, unsigned short* __restrict__ dst) {
    int i = blockIdx.x * blockDim.x + threadIdx.x; // over CP*HH
    if (i >= CP * HH) return;
    int n = i >> 11;        // / HH
    int k = i & (HH - 1);
    dst[i] = (n < CC) ? f2bf(w2[n * HH + k]) : (unsigned short)0;
}

__global__ void pad_b2(const float* __restrict__ b2, float* __restrict__ dst) {
    int i = blockIdx.x * blockDim.x + threadIdx.x;
    if (i < CP) dst[i] = (i < CC) ? b2[i] : 0.0f;
}

// ---------------------------------------------------------------- segment mean
__global__ void segsum_kernel(const float* __restrict__ var_flat, const int* __restrict__ seg,
                              float* __restrict__ sums, float* __restrict__ counts) {
    long long i = (long long)blockIdx.x * blockDim.x + threadIdx.x; // over TT*FF
    if (i >= (long long)TT * FF) return;
    int t = (int)(i >> 5);
    int f = (int)(i & 31);
    int s = seg[t];
    atomicAdd(&sums[(long long)s * FF + f], var_flat[i]);
    if (f == 0) atomicAdd(&counts[s], 1.0f);
}

__global__ void means_kernel(const float* __restrict__ sums, const float* __restrict__ counts,
                             unsigned short* __restrict__ means_bf) {
    int i = blockIdx.x * blockDim.x + threadIdx.x; // over BB*FF
    if (i >= BB * FF) return;
    float c = fmaxf(counts[i >> 5], 1.0f);
    means_bf[i] = f2bf(sums[i] / c);
}

// ---------------------------------------------------------------- WMMA GEMM
// Out[M,N] = A[M,K](bf16) @ Bw[N,K](bf16)^T (+bias[n]) (+addend[m,n]) (relu?)
//
// Block = 256 threads (8 waves, 2x4), output tile 128x128, BLK_K = 32.
// A/B tiles double-buffered in LDS; filled with CDNA5 async global->LDS
// copies (ASYNCcnt) so the next tile streams while WMMAs consume this one.
// LDS row stride = 40 shorts (80B): lane stride of 20 words mod 64 banks
// visits 16 distinct banks -> conflict-free ds_load_b128 fragment reads.
// M,N multiples of 128 and K multiple of 32 -> no divergence, EXEC all-ones
// at every WMMA as the ISA requires.
#define LDA 40
__global__ void __launch_bounds__(256)
gemm_wmma_bf16(const unsigned short* __restrict__ A,
               const unsigned short* __restrict__ Bw,
               const float* __restrict__ bias,
               const float* __restrict__ addend,
               float* __restrict__ outF,
               unsigned short* __restrict__ outBf,
               int M, int N, int K, int relu) {
    __shared__ unsigned short Atile[2][128 * LDA];
    __shared__ unsigned short Btile[2][128 * LDA];

    const int tid  = threadIdx.x;
    const int lane = tid & 31;
    const int wave = tid >> 5;
    const int wm   = wave >> 2;      // 0..1  -> 64-row strip
    const int wn   = wave & 3;       // 0..3  -> 32-col strip
    const int kh   = lane >> 4;      // K-half held by this lane
    const int l15  = lane & 15;

    const int blockM = blockIdx.y * 128;
    const int blockN = blockIdx.x * 128;

    // tile-copy mapping: each thread owns 32B of A and 32B of B per K-step
    const int cr = tid >> 1;         // row 0..127
    const int ch = (tid & 1) * 16;   // K-offset 0 or 16

    const int nk = K >> 5;           // K / 32

    v8f acc[4][2] = {};

    // ---- preload tile 0
    {
        const unsigned short* ga = A + (size_t)(blockM + cr) * K + ch;
        const unsigned short* gb = Bw + (size_t)(blockN + cr) * K + ch;
        unsigned short* la = &Atile[0][cr * LDA + ch];
        unsigned short* lb = &Btile[0][cr * LDA + ch];
        cp16(la, ga); cp16(la + 8, ga + 8);
        cp16(lb, gb); cp16(lb + 8, gb + 8);
    }
    wait_async_copies();
    __syncthreads();

    for (int kt = 0; kt < nk; ++kt) {
        const int cur = kt & 1;
        // ---- prefetch next tile into the other buffer (async, overlaps WMMA)
        if (kt + 1 < nk) {
            const int nxt = cur ^ 1;
            const int k0n = (kt + 1) << 5;
            const unsigned short* ga = A + (size_t)(blockM + cr) * K + k0n + ch;
            const unsigned short* gb = Bw + (size_t)(blockN + cr) * K + k0n + ch;
            unsigned short* la = &Atile[nxt][cr * LDA + ch];
            unsigned short* lb = &Btile[nxt][cr * LDA + ch];
            cp16(la, ga); cp16(la + 8, ga + 8);
            cp16(lb, gb); cp16(lb + 8, gb + 8);
        }

        // ---- fragments from LDS (ds_load_b128, conflict-free)
        union U { v16bf v; uint4 q[2]; };
        U af[4];
#pragma unroll
        for (int mt = 0; mt < 4; ++mt) {
            int m = wm * 64 + mt * 16 + l15;
            const unsigned short* ap = &Atile[cur][m * LDA + kh * 8];
            af[mt].q[0] = *(const uint4*)(ap);
            af[mt].q[1] = *(const uint4*)(ap + 16);
        }
        U bfr[2];
#pragma unroll
        for (int nt = 0; nt < 2; ++nt) {
            int n = wn * 32 + nt * 16 + l15;
            const unsigned short* bp = &Btile[cur][n * LDA + kh * 8];
            bfr[nt].q[0] = *(const uint4*)(bp);
            bfr[nt].q[1] = *(const uint4*)(bp + 16);
        }
#pragma unroll
        for (int mt = 0; mt < 4; ++mt)
#pragma unroll
            for (int nt = 0; nt < 2; ++nt)
                acc[mt][nt] = __builtin_amdgcn_wmma_f32_16x16x32_bf16(
                    false, af[mt].v, false, bfr[nt].v, (short)0, acc[mt][nt], false, false);

        // next-tile data must be resident and this tile's readers done
        wait_async_copies();
        __syncthreads();
    }

    // ---- epilogue: D layout -> m = r + 8*kh, n = lane&15 per 16x16 tile
#pragma unroll
    for (int mt = 0; mt < 4; ++mt) {
#pragma unroll
        for (int r = 0; r < 8; ++r) {
            int gm = blockM + wm * 64 + mt * 16 + r + 8 * kh;
#pragma unroll
            for (int nt = 0; nt < 2; ++nt) {
                int gn = blockN + wn * 32 + nt * 16 + l15;
                float v = acc[mt][nt][r];
                if (bias)   v += bias[gn];
                if (addend) v += addend[(size_t)gm * N + gn];
                if (relu)   v = fmaxf(v, 0.0f);
                size_t o = (size_t)gm * N + gn;
                if (outF)  outF[o]  = v;
                if (outBf) outBf[o] = f2bf(v);
            }
        }
    }
}

// ---------------------------------------------------------------- scalar head
__global__ void scalar_head(const unsigned short* __restrict__ hidden,
                            const float* __restrict__ Ws, const float* __restrict__ bs,
                            float* __restrict__ scalar_out) {
    __shared__ float red[256];
    int b = blockIdx.x;
    float s = 0.0f;
    for (int k = threadIdx.x; k < HH; k += 256)
        s += bf2f(hidden[(size_t)b * HH + k]) * Ws[k];
    red[threadIdx.x] = s;
    __syncthreads();
    for (int st = 128; st > 0; st >>= 1) {
        if (threadIdx.x < st) red[threadIdx.x] += red[threadIdx.x + st];
        __syncthreads();
    }
    if (threadIdx.x == 0) scalar_out[b] = red[0] + bs[0];
}

// ---------------------------------------------------------------- loss
// acc[0] += -logp_target, acc[1] += (scalar-target)^2, acc[2] += correct
__global__ void loss_kernel(const float* __restrict__ vecout,
                            const int* __restrict__ target_vector,
                            const float* __restrict__ scalar_out,
                            const float* __restrict__ target_scalar,
                            float* __restrict__ accs) {
    __shared__ float smax[256];
    __shared__ int   sidx[256];
    __shared__ float ssum[256];
    int b = blockIdx.x;
    const float* row = vecout + (size_t)b * CP;

    float lmax = -INFINITY; int lidx = 0;
    for (int n = threadIdx.x; n < CC; n += 256) {
        float v = row[n];
        if (v > lmax) { lmax = v; lidx = n; }
    }
    smax[threadIdx.x] = lmax; sidx[threadIdx.x] = lidx;
    __syncthreads();
    for (int st = 128; st > 0; st >>= 1) {
        if (threadIdx.x < st) {
            float a = smax[threadIdx.x], c = smax[threadIdx.x + st];
            int ai = sidx[threadIdx.x], ci = sidx[threadIdx.x + st];
            if (c > a || (c == a && ci < ai)) { smax[threadIdx.x] = c; sidx[threadIdx.x] = ci; }
        }
        __syncthreads();
    }
    float mx = smax[0];
    int amax = sidx[0];
    __syncthreads();

    float se = 0.0f;
    for (int n = threadIdx.x; n < CC; n += 256) se += expf(row[n] - mx);
    ssum[threadIdx.x] = se;
    __syncthreads();
    for (int st = 128; st > 0; st >>= 1) {
        if (threadIdx.x < st) ssum[threadIdx.x] += ssum[threadIdx.x + st];
        __syncthreads();
    }

    if (threadIdx.x == 0) {
        int tv = target_vector[b];
        float logp = row[tv] - mx - logf(ssum[0]);
        atomicAdd(&accs[0], -logp);
        float d = scalar_out[b] - target_scalar[b];
        atomicAdd(&accs[1], d * d);
        if (amax == tv) atomicAdd(&accs[2], 1.0f);
    }
}

__global__ void finalize_kernel(const float* __restrict__ accs, float* __restrict__ out) {
    if (threadIdx.x == 0 && blockIdx.x == 0) {
        const float invB = 1.0f / (float)BB;
        float lv = accs[0] * invB;
        float ls = accs[1] * invB;
        float ac = accs[2] * invB;
        out[0] = lv + ls;
        out[1] = lv;
        out[2] = ls;
        out[3] = ac;
    }
}

// ---------------------------------------------------------------- launch
static inline size_t alignUp(size_t x) { return (x + 255) & ~(size_t)255; }

extern "C" void kernel_launch(void* const* d_in, const int* in_sizes, int n_in,
                              void* d_out, int out_size, void* d_ws, size_t ws_size,
                              hipStream_t stream) {
    const float* feature_vector = (const float*)d_in[0];
    const float* var_flat       = (const float*)d_in[1];
    const int*   segment_ids    = (const int*)d_in[2];
    const int*   target_vector  = (const int*)d_in[3];
    const float* target_scalar  = (const float*)d_in[4];
    const float* Wv             = (const float*)d_in[5];
    const float* bv             = (const float*)d_in[6];
    const float* W1             = (const float*)d_in[7];
    const float* b1             = (const float*)d_in[8];
    const float* W2             = (const float*)d_in[9];
    const float* b2             = (const float*)d_in[10];
    const float* Ws             = (const float*)d_in[11];
    const float* bs             = (const float*)d_in[12];
    float* out = (float*)d_out;

    // workspace carve-up (contiguous zero region first: sums, counts, accs)
    char* w = (char*)d_ws;
    size_t off = 0;
    float* sums   = (float*)(w + off); off = alignUp(off + (size_t)BB * FF * 4);
    float* counts = (float*)(w + off); off = alignUp(off + (size_t)BB * 4);
    float* accs   = (float*)(w + off); off = alignUp(off + 64 * 4);
    size_t zeroFloats = ((size_t)((char*)accs - (char*)sums) / 4) + 64;
    unsigned short* means_bf = (unsigned short*)(w + off); off = alignUp(off + (size_t)BB * FF * 2);
    unsigned short* Wv_bf    = (unsigned short*)(w + off); off = alignUp(off + (size_t)DD * FF * 2);
    unsigned short* W1_bf    = (unsigned short*)(w + off); off = alignUp(off + (size_t)HH * DD * 2);
    unsigned short* W2p_bf   = (unsigned short*)(w + off); off = alignUp(off + (size_t)CP * HH * 2);
    float* b2p               = (float*)(w + off);          off = alignUp(off + (size_t)CP * 4);
    unsigned short* comb_bf  = (unsigned short*)(w + off); off = alignUp(off + (size_t)BB * DD * 2);
    unsigned short* hid_bf   = (unsigned short*)(w + off); off = alignUp(off + (size_t)BB * HH * 2);
    float* vecout            = (float*)(w + off);          off = alignUp(off + (size_t)BB * CP * 4);
    float* scalar_out        = (float*)(w + off);          off = alignUp(off + (size_t)BB * 4);
    (void)ws_size; (void)in_sizes; (void)n_in; (void)out_size;

    // 1) zero accumulators
    zero_f32<<<(int)((zeroFloats + 255) / 256), 256, 0, stream>>>(sums, (int)zeroFloats);

    // 2) segment sums + counts (pure HBM stream: ~105MB -> ~4.5us at 23.3TB/s)
    {
        long long n = (long long)TT * FF;
        segsum_kernel<<<(int)((n + 255) / 256), 256, 0, stream>>>(var_flat, segment_ids, sums, counts);
    }
    means_kernel<<<(BB * FF + 255) / 256, 256, 0, stream>>>(sums, counts, means_bf);

    // 3) weight conversions (bf16, padded where needed)
    cvt_f32_bf16<<<(DD * FF + 255) / 256, 256, 0, stream>>>(Wv, Wv_bf, DD * FF);
    cvt_f32_bf16<<<(HH * DD + 255) / 256, 256, 0, stream>>>(W1, W1_bf, HH * DD);
    cvt_w2_pad<<<(CP * HH + 255) / 256, 256, 0, stream>>>(W2, W2p_bf);
    pad_b2<<<(CP + 255) / 256, 256, 0, stream>>>(b2, b2p);

    // 4) GEMM1: combined = feature + means @ Wv^T + bv   [4096 x 1024], K=32
    {
        dim3 grid(DD / 128, BB / 128);
        gemm_wmma_bf16<<<grid, 256, 0, stream>>>(means_bf, Wv_bf, bv, feature_vector,
                                                 nullptr, comb_bf, BB, DD, FF, 0);
    }
    // 5) GEMM2: hidden = relu(combined @ W1^T + b1)      [4096 x 2048], K=1024
    {
        dim3 grid(HH / 128, BB / 128);
        gemm_wmma_bf16<<<grid, 256, 0, stream>>>(comb_bf, W1_bf, b1, nullptr,
                                                 nullptr, hid_bf, BB, HH, DD, 1);
    }
    // 6) GEMM3: vecout = hidden @ W2p^T + b2p            [4096 x 1024pad], K=2048
    {
        dim3 grid(CP / 128, BB / 128);
        gemm_wmma_bf16<<<grid, 256, 0, stream>>>(hid_bf, W2p_bf, b2p, nullptr,
                                                 vecout, nullptr, BB, CP, HH, 0);
    }

    // 7) scalar head, losses, finalize
    scalar_head<<<BB, 256, 0, stream>>>(hid_bf, Ws, bs, scalar_out);
    loss_kernel<<<BB, 256, 0, stream>>>(vecout, target_vector, scalar_out, target_scalar, accs);
    finalize_kernel<<<1, 64, 0, stream>>>(accs, out);
}
